// MultiheadAttention_49563922596294
// MI455X (gfx1250) — matrix operand order repaired and checked by reading, not compiled, not run on hardware
//
#include <hip/hip_runtime.h>

typedef __bf16 bf16;
typedef __bf16 bf16x8 __attribute__((ext_vector_type(8)));
typedef __bf16 v16bf  __attribute__((ext_vector_type(16)));
typedef float  v8f    __attribute__((ext_vector_type(8)));
typedef int    v4i    __attribute__((ext_vector_type(4)));

#define D_MODEL 2048
#define N_HEADS 16
#define DH 128
#define SEQ 1024
#define BATCH 4
#define THREE_D 6144
#define NTOK 4096
#define CLIP_QKV 8.0f
#define SOFTMAX_SCALE 0.08838834764831845f  // 1/sqrt(128)
#define LN_EPS 1e-5f
#define NEG_INF (-3.0e38f)

// ---------------- helpers ----------------

#ifndef __has_builtin
#define __has_builtin(x) 0
#endif

#if __has_builtin(__builtin_amdgcn_global_load_async_to_lds_b128)
#define HAVE_ASYNC_LDS 1
#define AS1 __attribute__((address_space(1)))
#define AS3 __attribute__((address_space(3)))
// async 16-byte copy global -> LDS (ASYNCcnt tracked, no VGPR round-trip)
__device__ __forceinline__ void copy16(const bf16* g, bf16* l) {
  __builtin_amdgcn_global_load_async_to_lds_b128(
      (AS1 v4i*)(AS1 void*)(void*)const_cast<bf16*>(g),
      (AS3 v4i*)(AS3 void*)(void*)l, 0, 0);
}
__device__ __forceinline__ void wait_async() {
  asm volatile("s_wait_asynccnt 0" ::: "memory");
}
#else
__device__ __forceinline__ void copy16(const bf16* g, bf16* l) {
  *(bf16x8*)l = *(const bf16x8*)g;
}
__device__ __forceinline__ void wait_async() {}
#endif

__device__ __forceinline__ bf16 f2bf(float f) {
  unsigned int u = __builtin_bit_cast(unsigned int, f);
  unsigned int lsb = (u >> 16) & 1u;
  u += 0x7fffu + lsb;                 // round-to-nearest-even
  unsigned short h = (unsigned short)(u >> 16);
  return __builtin_bit_cast(bf16, h);
}

__device__ __forceinline__ v8f wmma_bf16(v16bf a, v16bf b, v8f c) {
  return __builtin_amdgcn_wmma_f32_16x16x32_bf16(
      false, a, false, b, (short)0, c, false, false);
}

// A fragment 16x32 (MxK), bf16. Per ISA table:
// lanes 0-15 : row=lane,    elems 0-7 = K0..7,  elems 8-15 = K16..23
// lanes 16-31: row=lane-16, elems 0-7 = K8..15, elems 8-15 = K24..31
__device__ __forceinline__ v16bf load_a_frag(const bf16* base, int ld) {
  const int lane = threadIdx.x & 31;
  const int row  = lane & 15;
  const int k0   = (lane & 16) ? 8 : 0;
  const bf16* p = base + row * ld + k0;
  bf16x8 lo = *(const bf16x8*)(p);
  bf16x8 hi = *(const bf16x8*)(p + 16);
  v16bf r;
#pragma unroll
  for (int i = 0; i < 8; ++i) { r[i] = lo[i]; r[i + 8] = hi[i]; }
  return r;
}

// B fragment 32x16 (KxN), source stored as [N][K] row-major (K contiguous).
// lanes 0-15 : col=lane,    K 0..7,16..23 ; lanes 16-31: col=lane-16, K 8..15,24..31
__device__ __forceinline__ v16bf load_b_frag(const bf16* base, int ld) {
  const int lane = threadIdx.x & 31;
  const int n    = lane & 15;
  const int k0   = (lane & 16) ? 8 : 0;
  const bf16* p = base + n * ld + k0;
  bf16x8 lo = *(const bf16x8*)(p);
  bf16x8 hi = *(const bf16x8*)(p + 16);
  v16bf r;
#pragma unroll
  for (int i = 0; i < 8; ++i) { r[i] = lo[i]; r[i + 8] = hi[i]; }
  return r;
}

// ---------------- kernel 0: fp32 -> bf16 convert ----------------

__global__ __launch_bounds__(256) void f32_to_bf16_kernel(
    const float* __restrict__ in, bf16* __restrict__ out, int n4) {
  int i = blockIdx.x * 256 + threadIdx.x;
  if (i < n4) {
    float4 v = ((const float4*)in)[i];
    bf16* o = out + (size_t)i * 4;
    o[0] = f2bf(v.x); o[1] = f2bf(v.y); o[2] = f2bf(v.z); o[3] = f2bf(v.w);
  }
}

// ---------------- kernel 1/4: NT GEMM  C[M,N] = A[M,K] * B[N,K]^T + bias --
// 128x128 tile, double-buffered LDS, one barrier per K-step of 32.

template <bool CLAMP>
__global__ __launch_bounds__(256) void gemm_nt_bf16(
    const bf16* __restrict__ A, const bf16* __restrict__ Bm,
    const float* __restrict__ bias, float* __restrict__ C,
    int M, int N, int K) {
  (void)M;
  __shared__ bf16 As[2][128 * 40];
  __shared__ bf16 Bs[2][128 * 40];

  const int t    = threadIdx.x;
  const int wid  = t >> 5;
  const int lane = t & 31;
  const int wm = wid & 3;        // 4 waves along M (32 rows each)
  const int wn = wid >> 2;       // 2 waves along N (64 cols each)
  const int m0 = blockIdx.y * 128;
  const int n0 = blockIdx.x * 128;

  v8f acc[2][4] = {};

  // tile fill: 128 rows x 32 elems = 512 chunks of 8 elems (16B); 2 per thread
  auto issue_tile = [&](int kt, int bufi) {
#pragma unroll
    for (int j = 0; j < 2; ++j) {
      int idx = t + j * 256;           // 0..511
      int r = idx >> 2;                // row 0..127
      int c = (idx & 3) * 8;           // 8-elem chunk
      const bf16* ga = A  + (size_t)(m0 + r) * K + kt + c;
      const bf16* gb = Bm + (size_t)(n0 + r) * K + kt + c;
      copy16(ga, &As[bufi][r * 40 + c]);
      copy16(gb, &Bs[bufi][r * 40 + c]);
      if (kt + 64 < K) {               // distance-2 prefetch into L2/WGP$
        __builtin_prefetch(ga + 64, 0, 3);
        __builtin_prefetch(gb + 64, 0, 3);
      }
    }
  };

  issue_tile(0, 0);
  const int nk = K >> 5;
  for (int i = 0; i < nk; ++i) {
    wait_async();
    __syncthreads();                   // buf[i&1] fully resident for all waves
    if (i + 1 < nk) issue_tile((i + 1) * 32, (i + 1) & 1);

    const bf16* as = As[i & 1];
    const bf16* bs = Bs[i & 1];
    v16bf af0 = load_a_frag(as + (wm * 32 + 0) * 40, 40);
    v16bf af1 = load_a_frag(as + (wm * 32 + 16) * 40, 40);
#pragma unroll
    for (int nj = 0; nj < 4; ++nj) {
      v16bf bf = load_b_frag(bs + (wn * 64 + nj * 16) * 40, 40);
      acc[0][nj] = wmma_bf16(af0, bf, acc[0][nj]);
      acc[1][nj] = wmma_bf16(af1, bf, acc[1][nj]);
    }
  }

  // epilogue (C-layout: lane = col, vgpr j = row; lanes>=16 -> row+8)
  const int coll   = lane & 15;
  const int rowadd = (lane & 16) ? 8 : 0;
#pragma unroll
  for (int mi = 0; mi < 2; ++mi) {
#pragma unroll
    for (int nj = 0; nj < 4; ++nj) {
      int col = n0 + wn * 64 + nj * 16 + coll;
      float bv = bias[col];
#pragma unroll
      for (int j = 0; j < 8; ++j) {
        int row = m0 + wm * 32 + mi * 16 + j + rowadd;
        float v = acc[mi][nj][j] + bv;
        if (CLAMP) v = fminf(fmaxf(v, -CLIP_QKV), CLIP_QKV);
        C[(size_t)row * N + col] = v;
      }
    }
  }
}

// ---------------- kernel 2: QK layernorm + repack -------------------------
// Q -> [B,H,S,Dh] bf16, PRE-SCALED by SOFTMAX_SCALE
// K -> [B,H,S,Dh] bf16 ;  V -> [B,H,Dh,S] bf16 (pre-transposed for PV gemm)

__device__ __forceinline__ float block_sum(float v, float* red) {
  int t = threadIdx.x;
  red[t] = v; __syncthreads();
  for (int off = 128; off > 0; off >>= 1) {
    if (t < off) red[t] += red[t + off];
    __syncthreads();
  }
  float r = red[0]; __syncthreads();
  return r;
}

__global__ __launch_bounds__(256) void qk_ln_pack(
    const float* __restrict__ qkv,
    const float* __restrict__ qg, const float* __restrict__ qb,
    const float* __restrict__ kg, const float* __restrict__ kb,
    bf16* __restrict__ Qo, bf16* __restrict__ Ko, bf16* __restrict__ Vo) {
  __shared__ float red[256];
  const int row = blockIdx.x;          // token index
  const int b = row >> 10, s = row & 1023;
  const int t = threadIdx.x;
  const float* qr = qkv + (size_t)row * THREE_D;
  const float* kr = qr + D_MODEL;
  const float* vr = qr + 2 * D_MODEL;

  float qv[8], kv[8];
  float qs = 0.f, qss = 0.f, ks = 0.f, kss = 0.f;
#pragma unroll
  for (int i = 0; i < 8; ++i) {
    int d = i * 256 + t;
    float q = qr[d], k = kr[d];
    qv[i] = q; kv[i] = k;
    qs += q; qss += q * q; ks += k; kss += k * k;
  }
  float qsum = block_sum(qs, red);
  float qsq  = block_sum(qss, red);
  float ksum = block_sum(ks, red);
  float ksq  = block_sum(kss, red);
  const float inv = 1.0f / (float)D_MODEL;
  float qmu = qsum * inv, kmu = ksum * inv;
  float qrs = rsqrtf(qsq * inv - qmu * qmu + LN_EPS);
  float krs = rsqrtf(ksq * inv - kmu * kmu + LN_EPS);

#pragma unroll
  for (int i = 0; i < 8; ++i) {
    int d = i * 256 + t;
    int h = d >> 7, dh = d & 127;
    size_t bh = (size_t)(b * N_HEADS + h);
    size_t dqk = (bh * SEQ + s) * DH + dh;
    // fold softmax scale into Q:  (q*s)·k == (q·k)*s
    Qo[dqk] = f2bf(((qv[i] - qmu) * qrs * qg[d] + qb[d]) * SOFTMAX_SCALE);
    Ko[dqk] = f2bf((kv[i] - kmu) * krs * kg[d] + kb[d]);
    Vo[(bh * DH + dh) * SEQ + s] = f2bf(vr[d]);   // transposed
  }
}

// ---------------- kernel 3: flash attention -------------------------------

#define SDH 136   // Ks row stride (elems)
#define SVT 72    // Vt / Ps row stride (elems)

__global__ __launch_bounds__(128) void flash_attn(
    const bf16* __restrict__ Q, const bf16* __restrict__ Kg,
    const bf16* __restrict__ Vg, const float* __restrict__ abias,
    const unsigned char* __restrict__ amask, bf16* __restrict__ Ctx) {
  __shared__ bf16 Ks[64 * SDH];       // [key][dh]
  __shared__ bf16 Vt[128 * SVT];      // [dh][key]  (already transposed in mem)
  __shared__ bf16 Ps[4 * 16 * SVT];   // per-wave P tiles [q][key]

  const int t = threadIdx.x;
  const int lane = t & 31;
  const int wq = t >> 5;              // wave -> 16-query slice
  const int bh = blockIdx.x;
  const int b = bh >> 4, h = bh & 15;
  const int q0 = blockIdx.y * 64;
  const size_t baseBH = (size_t)bh * SEQ * DH;   // for Q,K ([B,H,S,Dh])
  const size_t vbase  = (size_t)bh * DH * SEQ;   // for V ([B,H,Dh,S])

  // preload Q fragments (4 K-steps over Dh=128)
  v16bf aq[4];
  const bf16* qbase = Q + baseBH + (size_t)(q0 + wq * 16) * DH;
#pragma unroll
  for (int kt = 0; kt < 4; ++kt) aq[kt] = load_a_frag(qbase + kt * 32, DH);

  v8f o[8] = {};
  float m8[8], l8[8];
#pragma unroll
  for (int j = 0; j < 8; ++j) { m8[j] = NEG_INF; l8[j] = 0.f; }

  const int coll   = lane & 15;
  const int rowadd = (lane & 16) ? 8 : 0;

  // one key-block step; `causal` is compile-time constant at each call site
  auto process_tile = [&](int kbase, bool causal) {
    // K tile: 64x128, V tile: 128x64 -> 1024 16B chunks each side, 8/thread
#pragma unroll
    for (int i = 0; i < 8; ++i) {
      int idx = t + i * 128;          // 0..1023
      int rk = idx >> 4;              // key 0..63
      int ck = (idx & 15) * 8;        // dh chunk of 8
      copy16(Kg + baseBH + (size_t)(kbase + rk) * DH + ck,
             &Ks[rk * SDH + ck]);
      int rv = idx >> 3;              // dh 0..127
      int cv = (idx & 7) * 8;         // key chunk of 8
      copy16(Vg + vbase + (size_t)rv * SEQ + kbase + cv,
             &Vt[rv * SVT + cv]);
    }
    wait_async();
    __syncthreads();

    // scores: S = Qs * K^T  (16 queries x 64 keys per wave; scale pre-folded)
    v8f sf[4] = {};
#pragma unroll
    for (int nf = 0; nf < 4; ++nf) {
#pragma unroll
      for (int kt = 0; kt < 4; ++kt) {
        v16bf bk = load_b_frag(Ks + (nf * 16) * SDH + kt * 32, SDH);
        sf[nf] = wmma_bf16(aq[kt], bk, sf[nf]);
      }
    }

    // bias (padding mask folded in) + optional causal mask + online softmax
    float pbuf[4][8];
    float tmax[8];
#pragma unroll
    for (int j = 0; j < 8; ++j) tmax[j] = NEG_INF;
#pragma unroll
    for (int nf = 0; nf < 4; ++nf) {
      int kc = kbase + nf * 16 + coll;
      float bia = (amask[b * SEQ + kc] != 0) ? abias[h * SEQ + kc] : NEG_INF;
#pragma unroll
      for (int j = 0; j < 8; ++j) {
        float s = sf[nf][j] + bia;
        if (causal) {                  // only diagonal tile pays for this
          int qi = q0 + wq * 16 + j + rowadd;
          if (kc > qi) s = NEG_INF;
        }
        pbuf[nf][j] = s;
        tmax[j] = fmaxf(tmax[j], s);
      }
    }
#pragma unroll
    for (int j = 0; j < 8; ++j) {
#pragma unroll
      for (int off = 1; off < 16; off <<= 1)
        tmax[j] = fmaxf(tmax[j], __shfl_xor(tmax[j], off, 32));
    }
    float corr[8];
#pragma unroll
    for (int j = 0; j < 8; ++j) {
      float mn = fmaxf(m8[j], tmax[j]);
      corr[j] = __expf(m8[j] - mn);
      m8[j] = mn;
      l8[j] *= corr[j];
    }
    float rsum[8] = {};
#pragma unroll
    for (int nf = 0; nf < 4; ++nf) {
#pragma unroll
      for (int j = 0; j < 8; ++j) {
        float p = __expf(pbuf[nf][j] - m8[j]);
        rsum[j] += p;
        Ps[(wq * 16 + j + rowadd) * SVT + nf * 16 + coll] = f2bf(p);
      }
    }
#pragma unroll
    for (int j = 0; j < 8; ++j) {
#pragma unroll
      for (int off = 1; off < 16; off <<= 1)
        rsum[j] += __shfl_xor(rsum[j], off, 32);
      l8[j] += rsum[j];
    }
#pragma unroll
    for (int df = 0; df < 8; ++df)
#pragma unroll
      for (int j = 0; j < 8; ++j) o[df][j] *= corr[j];

    asm volatile("s_wait_dscnt 0" ::: "memory");  // P visible to own wave

    // context accumulation: O += P * V
#pragma unroll
    for (int kk = 0; kk < 2; ++kk) {
      v16bf ap = load_a_frag(Ps + (wq * 16) * SVT + kk * 32, SVT);
#pragma unroll
      for (int df = 0; df < 8; ++df) {
        v16bf bv = load_b_frag(Vt + (df * 16) * SVT + kk * 32, SVT);
        o[df] = wmma_bf16(ap, bv, o[df]);
      }
    }
    __syncthreads();
  };

  const int ndiag = blockIdx.y;       // full (unmasked) tiles
  for (int ib = 0; ib < ndiag; ++ib) process_tile(ib * 64, false);
  process_tile(ndiag * 64, true);     // diagonal tile: causal masking

  // normalize and store ctx in [B,S,D] bf16
  float rl[8];
#pragma unroll
  for (int j = 0; j < 8; ++j) rl[j] = 1.0f / l8[j];
#pragma unroll
  for (int df = 0; df < 8; ++df) {
#pragma unroll
    for (int j = 0; j < 8; ++j) {
      int qi = q0 + wq * 16 + j + rowadd;
      size_t dst = ((size_t)(b * SEQ + qi)) * D_MODEL + h * DH + df * 16 + coll;
      Ctx[dst] = f2bf(o[df][j] * rl[j]);
    }
  }
}

// ---------------- host launch ---------------------------------------------

extern "C" void kernel_launch(void* const* d_in, const int* in_sizes, int n_in,
                              void* d_out, int out_size, void* d_ws, size_t ws_size,
                              hipStream_t stream) {
  (void)in_sizes; (void)n_in; (void)out_size; (void)ws_size;
  const float* x       = (const float*)d_in[0];
  const float* abias   = (const float*)d_in[1];
  const unsigned char* amask = (const unsigned char*)d_in[2];
  const float* Wqkv_w  = (const float*)d_in[3];
  const float* Wqkv_b  = (const float*)d_in[4];
  const float* q_g     = (const float*)d_in[5];
  const float* q_b     = (const float*)d_in[6];
  const float* k_g     = (const float*)d_in[7];
  const float* k_b     = (const float*)d_in[8];
  const float* out_w   = (const float*)d_in[9];
  const float* out_b   = (const float*)d_in[10];
  float* out = (float*)d_out;

  char* ws = (char*)d_ws;
  bf16* xb    = (bf16*)ws;  ws += (size_t)NTOK * D_MODEL * 2;       // 16.8 MB
  bf16* wqkvb = (bf16*)ws;  ws += (size_t)THREE_D * D_MODEL * 2;    // 25.2 MB
  bf16* outwb = (bf16*)ws;  ws += (size_t)D_MODEL * D_MODEL * 2;    //  8.4 MB
  float* qkv  = (float*)ws; ws += (size_t)NTOK * THREE_D * 4;       // 100.7 MB
  bf16* Qb    = (bf16*)ws;  ws += (size_t)NTOK * D_MODEL * 2;
  bf16* Kb    = (bf16*)ws;  ws += (size_t)NTOK * D_MODEL * 2;
  bf16* Vb    = (bf16*)ws;  ws += (size_t)NTOK * D_MODEL * 2;       // [B,H,Dh,S]
  bf16* Ctxb  = (bf16*)ws;  ws += (size_t)NTOK * D_MODEL * 2;

  const int nx = NTOK * D_MODEL;          // 8,388,608
  const int nw = THREE_D * D_MODEL;       // 12,582,912
  const int no = D_MODEL * D_MODEL;       // 4,194,304
  f32_to_bf16_kernel<<<(nx / 4 + 255) / 256, 256, 0, stream>>>(x, xb, nx / 4);
  f32_to_bf16_kernel<<<(nw / 4 + 255) / 256, 256, 0, stream>>>(Wqkv_w, wqkvb, nw / 4);
  f32_to_bf16_kernel<<<(no / 4 + 255) / 256, 256, 0, stream>>>(out_w, outwb, no / 4);

  // qkv = clip(x @ Wqkv^T + b)
  gemm_nt_bf16<true><<<dim3(THREE_D / 128, NTOK / 128), 256, 0, stream>>>(
      xb, wqkvb, Wqkv_b, qkv, NTOK, THREE_D, D_MODEL);

  // layernorm q,k over d_model; repack q,k -> [B,H,S,Dh], v -> [B,H,Dh,S]
  qk_ln_pack<<<NTOK, 256, 0, stream>>>(qkv, q_g, q_b, k_g, k_b, Qb, Kb, Vb);

  // attention
  flash_attn<<<dim3(BATCH * N_HEADS, SEQ / 64), 128, 0, stream>>>(
      Qb, Kb, Vb, abias, amask, Ctxb);

  // out = ctx @ out_w^T + out_b
  gemm_nt_bf16<false><<<dim3(D_MODEL / 128, NTOK / 128), 256, 0, stream>>>(
      Ctxb, outwb, out_b, out, NTOK, D_MODEL, D_MODEL);
}
